// QuantLinear_10514079941161
// MI455X (gfx1250) — compile-verified, matrix operand order checked by reading, not run on hardware
//
#include <hip/hip_runtime.h>

typedef __attribute__((ext_vector_type(16))) _Float16 v16h;
typedef __attribute__((ext_vector_type(8)))  _Float16 v8h;
typedef __attribute__((ext_vector_type(4)))  _Float16 v4h;
typedef __attribute__((ext_vector_type(8)))  float    v8f;
typedef __attribute__((ext_vector_type(4)))  float    v4f;

// Problem constants (reference): INF = K = 4096, OUTF = N = 4096, GROUPSIZE = 128
#define KDIM 4096
#define NDIM 4096
#define BM 128            // block M tile
#define BN 256            // block N tile
#define BK 64             // block K tile (divides GROUPSIZE -> group const per tile)
#define LDT 72            // padded LDS row stride in halves (144 B = 9*16 B)
#define XTILE (BM * LDT)  // halves per X buffer
#define WTILE (BN * LDT)  // halves per W buffer
#define SMEM_HALVES (2 * XTILE + 2 * WTILE)
#define SMEM_BYTES (SMEM_HALVES * 2)   // 110592 B dynamic LDS (320 KB/WGP available)

__global__ __launch_bounds__(256)
void gptq_wmma_f16(const float* __restrict__ x,
                   const int*   __restrict__ qweight,
                   const int*   __restrict__ qzeros,
                   const float* __restrict__ scales,
                   const float* __restrict__ bias,
                   float*       __restrict__ out) {
    extern __shared__ __align__(16) _Float16 smem[];
    // layout: [X buf0][X buf1][W buf0][W buf1]

    const int tid  = threadIdx.x;
    const int lane = tid & 31;
    const int wave = tid >> 5;
    const int wm   = wave >> 2;        // 0..1 : 64 M rows per wave
    const int wn   = wave & 3;         // 0..3 : 64 N cols per wave
    const int m0   = blockIdx.y * BM;
    const int n0   = blockIdx.x * BN;
    const int xrow = tid >> 4;         // 0..15 : X staging row phase
    const int xc4  = tid & 15;         // float4 column within 64-wide K tile
    const int gn   = n0 + tid;         // W staging: fixed output column per thread

    v8f acc[4][4];
    #pragma unroll
    for (int mt = 0; mt < 4; ++mt)
        #pragma unroll
        for (int nt = 0; nt < 4; ++nt)
            acc[mt][nt] = (v8f){0.f, 0.f, 0.f, 0.f, 0.f, 0.f, 0.f, 0.f};

    // ---------------- prologue: stage tile 0 into buffer 0 ----------------
    {
        _Float16* dX = smem;                 // X buf0
        _Float16* dW = smem + 2 * XTILE;     // W buf0
        #pragma unroll
        for (int i = 0; i < 8; ++i) {
            const int row = i * 16 + xrow;
            const v4f xv = *(const v4f*)(x + (size_t)(m0 + row) * KDIM + xc4 * 4);
            v4h hv;
            hv[0] = (_Float16)xv[0]; hv[1] = (_Float16)xv[1];
            hv[2] = (_Float16)xv[2]; hv[3] = (_Float16)xv[3];
            *(v4h*)&dX[row * LDT + xc4 * 4] = hv;
        }
        const float z = (float)(((qzeros[gn >> 3] >> ((gn & 7) * 4)) & 0xF) + 1);
        const float s = scales[gn];
        const _Float16 hp = (_Float16)(s * ( 1.0f - z));
        const _Float16 hn = (_Float16)(s * (-1.0f - z));
        #pragma unroll
        for (int i = 0; i < 8; ++i) {
            const int word = qweight[(size_t)i * NDIM + gn];
            v8h wv;
            #pragma unroll
            for (int j = 0; j < 8; ++j)
                wv[j] = ((word >> (4 * j)) & 0xF) ? hp : hn;
            *(v8h*)&dW[tid * LDT + i * 8] = wv;
        }
    }
    __syncthreads();

    // ---------------- main loop: double-buffered ----------------
    const int NITER = KDIM / BK;
    for (int iter = 0; iter < NITER; ++iter) {
        const int k0  = iter * BK;
        const int cur = iter & 1;
        const int nxt = cur ^ 1;
        const _Float16* sX = smem + cur * XTILE;
        const _Float16* sW = smem + 2 * XTILE + cur * WTILE;

        // (1) issue global loads for the NEXT tile (in flight across the WMMAs)
        v4f xr[8];
        int wword[8];
        _Float16 hp = (_Float16)0.f, hn = (_Float16)0.f;
        const bool has_next = (iter + 1 < NITER);
        if (has_next) {
            const int kn = k0 + BK;
            #pragma unroll
            for (int i = 0; i < 8; ++i) {
                const int row = i * 16 + xrow;
                xr[i] = *(const v4f*)(x + (size_t)(m0 + row) * KDIM + kn + xc4 * 4);
            }
            #pragma unroll
            for (int i = 0; i < 8; ++i)
                wword[i] = qweight[(size_t)(kn / 8 + i) * NDIM + gn];
            const int g = kn / 128;
            const float z = (float)(((qzeros[g * (NDIM / 8) + (gn >> 3)] >> ((gn & 7) * 4)) & 0xF) + 1);
            const float s = scales[g * NDIM + gn];
            hp = (_Float16)(s * ( 1.0f - z));
            hn = (_Float16)(s * (-1.0f - z));
        }

        // (2) compute from current buffers: 2 K-steps x 16 WMMA
        #pragma unroll
        for (int ks = 0; ks < 2; ++ks) {
            const int koff = ks * 32 + ((lane >> 4) << 3);  // lanes 0-15: +0, 16-31: +8
            union { v16h v; v8h h[2]; } b[4];
            #pragma unroll
            for (int nt = 0; nt < 4; ++nt) {
                const _Float16* p = &sW[(wn * 64 + nt * 16 + (lane & 15)) * LDT + koff];
                b[nt].h[0] = *(const v8h*)p;
                b[nt].h[1] = *(const v8h*)(p + 16);
            }
            #pragma unroll
            for (int mt = 0; mt < 4; ++mt) {
                union { v16h v; v8h h[2]; } a;
                const _Float16* p = &sX[(wm * 64 + mt * 16 + (lane & 15)) * LDT + koff];
                a.h[0] = *(const v8h*)p;
                a.h[1] = *(const v8h*)(p + 16);
                #pragma unroll
                for (int nt = 0; nt < 4; ++nt)
                    acc[mt][nt] = __builtin_amdgcn_wmma_f32_16x16x32_f16(
                        false, a.v, false, b[nt].v, (short)0, acc[mt][nt], false, false);
            }
        }

        // (3) convert + store next tile into the other buffers
        if (has_next) {
            _Float16* dX = smem + nxt * XTILE;
            _Float16* dW = smem + 2 * XTILE + nxt * WTILE;
            #pragma unroll
            for (int i = 0; i < 8; ++i) {
                v4h hv;
                hv[0] = (_Float16)xr[i][0]; hv[1] = (_Float16)xr[i][1];
                hv[2] = (_Float16)xr[i][2]; hv[3] = (_Float16)xr[i][3];
                *(v4h*)&dX[(i * 16 + xrow) * LDT + xc4 * 4] = hv;
            }
            #pragma unroll
            for (int i = 0; i < 8; ++i) {
                v8h wv;
                #pragma unroll
                for (int j = 0; j < 8; ++j)
                    wv[j] = ((wword[i] >> (4 * j)) & 0xF) ? hp : hn;
                *(v8h*)&dW[tid * LDT + i * 8] = wv;
            }
        }
        __syncthreads();
    }

    // ---------------- epilogue: bias add + non-temporal store ----------------
    // C layout: VGPR r -> M = r (lanes 0-15) / 8+r (lanes 16-31); N = lane%16
    #pragma unroll
    for (int mt = 0; mt < 4; ++mt) {
        const int rbase = m0 + wm * 64 + mt * 16 + ((lane >> 4) << 3);
        #pragma unroll
        for (int nt = 0; nt < 4; ++nt) {
            const int col = n0 + wn * 64 + nt * 16 + (lane & 15);
            const float bv = bias[col];
            #pragma unroll
            for (int r = 0; r < 8; ++r)
                __builtin_nontemporal_store(acc[mt][nt][r] + bv,
                                            &out[(size_t)(rbase + r) * NDIM + col]);
        }
    }
}

extern "C" void kernel_launch(void* const* d_in, const int* in_sizes, int n_in,
                              void* d_out, int out_size, void* d_ws, size_t ws_size,
                              hipStream_t stream) {
    const float* x       = (const float*)d_in[0];
    const int*   qweight = (const int*)  d_in[1];
    const int*   qzeros  = (const int*)  d_in[2];
    const float* scales  = (const float*)d_in[3];
    const float* bias    = (const float*)d_in[4];
    // d_in[5] = g_idx : implied by k/GROUPSIZE, unused
    float* out = (float*)d_out;

    const int M = in_sizes[0] / KDIM;       // 8192
    dim3 grid(NDIM / BN, M / BM);           // (16, 64)
    dim3 block(256);
    gptq_wmma_f16<<<grid, block, SMEM_BYTES, stream>>>(x, qweight, qzeros, scales, bias, out);
}
